// LocalEncoder_26182120636879
// MI455X (gfx1250) — matrix operand order repaired and checked by reading, not compiled, not run on hardware
//
#include <hip/hip_runtime.h>
#include <hip/hip_bf16.h>

// ---------------- problem constants (match reference) ----------------
#define B_SZ   32
#define N_PTS  2048
#define S_PTS  256
#define K1_NB  32
#define K2_NB  20
#define R_PN   (B_SZ * S_PTS * K1_NB)   // 262144
#define R_S    (B_SZ * S_PTS)           // 8192
#define R_G    (B_SZ * S_PTS * K2_NB)   // 163840
#define BN_EPS 1e-5f

typedef __attribute__((ext_vector_type(16))) _Float16 v16h;
typedef __attribute__((ext_vector_type(8)))  _Float16 v8h;
typedef __attribute__((ext_vector_type(8)))  float    v8f;

// ---------------------------------------------------------------------
// FPS: one block per batch. dist[] lives in LDS, 256 dependent argmax steps.
__global__ void __launch_bounds__(256)
fps_kernel(const float* __restrict__ xyz, int* __restrict__ fps_idx,
           float* __restrict__ nx_ws, float* __restrict__ nx_out)
{
    const int b   = blockIdx.x;
    const int tid = threadIdx.x;
    __shared__ float dist[N_PTS];
    __shared__ float rv[256];
    __shared__ int   ri[256];
    __shared__ float cpt[3];
    __shared__ int   far_s;

    const float* P = xyz + (size_t)b * N_PTS * 3;
    for (int i = tid; i < N_PTS; i += 256) dist[i] = 1e10f;
    if (tid == 0) far_s = 0;
    __syncthreads();

    for (int it = 0; it < S_PTS; ++it) {
        if (tid == 0) {
            int f = far_s;
            fps_idx[b * S_PTS + it] = f;
            float x = P[f * 3 + 0], y = P[f * 3 + 1], z = P[f * 3 + 2];
            cpt[0] = x; cpt[1] = y; cpt[2] = z;
            size_t o = ((size_t)b * S_PTS + it) * 3;
            nx_ws[o + 0] = x; nx_ws[o + 1] = y; nx_ws[o + 2] = z;
            nx_out[o + 0] = x; nx_out[o + 1] = y; nx_out[o + 2] = z;
        }
        __syncthreads();
        float cx = cpt[0], cy = cpt[1], cz = cpt[2];
        float best = -1.0f; int bi = 0;
        for (int i = tid; i < N_PTS; i += 256) {
            float dx = P[i * 3 + 0] - cx, dy = P[i * 3 + 1] - cy, dz = P[i * 3 + 2] - cz;
            float d  = dx * dx + dy * dy + dz * dz;
            float nd = fminf(dist[i], d);
            dist[i]  = nd;
            if (nd > best) { best = nd; bi = i; }   // ascending i -> smallest idx on tie
        }
        rv[tid] = best; ri[tid] = bi;
        __syncthreads();
        for (int s = 128; s > 0; s >>= 1) {
            if (tid < s) {
                float ov = rv[tid + s]; int oi = ri[tid + s];
                if (ov > rv[tid] || (ov == rv[tid] && oi < ri[tid])) { rv[tid] = ov; ri[tid] = oi; }
            }
            __syncthreads();
        }
        if (tid == 0) far_s = ri[0];
        __syncthreads();
    }
}

// ---------------------------------------------------------------------
// Brute-force KNN, one thread per query, insertion into sorted top-K.
template <int KK>
__global__ void __launch_bounds__(256)
knn_kernel(const float* __restrict__ pts, const float* __restrict__ qry,
           int* __restrict__ idx, int Np, int Sq)
{
    int t = blockIdx.x * blockDim.x + threadIdx.x;
    if (t >= B_SZ * Sq) return;
    int b = t / Sq;
    const float* P = pts + (size_t)b * Np * 3;
    const float* q = qry + (size_t)t * 3;
    float qx = q[0], qy = q[1], qz = q[2];
    float bd[KK]; int bi[KK];
#pragma unroll
    for (int k = 0; k < KK; ++k) { bd[k] = 3.0e38f; bi[k] = 0; }
    for (int i = 0; i < Np; ++i) {
        float dx = P[i * 3 + 0] - qx, dy = P[i * 3 + 1] - qy, dz = P[i * 3 + 2] - qz;
        float d  = dx * dx + dy * dy + dz * dz;
        if (d < bd[KK - 1]) {
            int j = KK - 1;
            while (j > 0 && bd[j - 1] > d) { bd[j] = bd[j - 1]; bi[j] = bi[j - 1]; --j; }
            bd[j] = d; bi[j] = i;
        }
    }
    for (int k = 0; k < KK; ++k) idx[(size_t)t * KK + k] = bi[k];
}

// ---------------------------------------------------------------------
// Convert + transpose weights: src (Cin x Cout) f32 -> dst (Cout x Cin) f16.
// Puts each output column's K-run contiguous so WMMA B-fragments load as b128s.
__global__ void cvt_transpose_kernel(const float* __restrict__ src, _Float16* __restrict__ dst,
                                     int Cin, int Cout)
{
    int t = blockIdx.x * blockDim.x + threadIdx.x;
    if (t >= Cin * Cout) return;
    int k = t / Cout, c = t % Cout;
    dst[(size_t)c * Cin + k] = (_Float16)src[t];
}

// ---------------------------------------------------------------------
// group + pn1 (3 -> 64) scalar GEMM, writes f16 Z.
__global__ void __launch_bounds__(256)
group_pn1_kernel(const float* __restrict__ xyz, const float* __restrict__ nx,
                 const int* __restrict__ knn, const float* __restrict__ w1,
                 _Float16* __restrict__ Z)
{
    int r = blockIdx.x * blockDim.x + threadIdx.x;   // (b*S+s)*K + k
    if (r >= R_PN) return;
    int bs = r / K1_NB;
    int b  = bs / S_PTS;
    int pid = knn[r];
    const float* P = xyz + ((size_t)b * N_PTS + pid) * 3;
    const float* C = nx + (size_t)bs * 3;
    float m0 = P[0] - C[0], m1 = P[1] - C[1], m2 = P[2] - C[2];
    _Float16* zr = Z + (size_t)r * 64;
#pragma unroll 8
    for (int c = 0; c < 64; ++c)
        zr[c] = (_Float16)(m0 * w1[c] + m1 * w1[64 + c] + m2 * w1[128 + c]);
}

// ---------------------------------------------------------------------
// WMMA GEMM: Z(R x COUT) = X(R x CIN) @ W(CIN x COUT), weights pre-transposed
// as Wt(COUT x CIN) f16 and staged through LDS (shared by the block's 8 waves).
// One wave owns a 16-row x COUT strip. Software-pipelined inner loop:
//   - A fragment for K-chunk k+1 prefetched into registers before the
//     COUT/16-WMMA burst of chunk k (hides global latency behind matrix ops)
//   - B fragments rotate through bcur/bnext so the ds_load for tile t+1 is
//     issued before the v_wmma consuming tile t.
// For CIN=288 the weight tile is K-sliced (96-deep, 24KB) with block barriers.
// LDS row pitch is padded +8 halfs to break the 256B bank-cycle stride.
template <int CIN, int COUT>
__global__ void __launch_bounds__(256)
wmma_gemm_f16(const _Float16* __restrict__ X, const _Float16* __restrict__ Wt,
              _Float16* __restrict__ Z, int R)
{
    constexpr int TN     = COUT / 16;
    constexpr int KS     = (CIN * COUT * 2 <= 49152) ? CIN : 96;   // K-slice depth
    constexpr int NSLICE = CIN / KS;
    constexpr int PITCH  = KS + 8;                                 // halfs, bank-depad
    __shared__ _Float16 Wl[PITCH * COUT];

    const int lane   = threadIdx.x & 31;
    const int wave   = blockIdx.x * 8 + (threadIdx.x >> 5);
    const int tilesM = R >> 4;
    const bool active = (wave < tilesM);       // wave-uniform predicate
    const int hi  = lane >> 4;                 // 0 or 1
    const int l16 = lane & 15;

    v8f acc[TN];
#pragma unroll
    for (int t = 0; t < TN; ++t) acc[t] = (v8f){};

    const _Float16* xrow = X + (size_t)((active ? wave : 0) * 16 + l16) * CIN;

    // A fragment: lanes 0-15 row M=l16, K=+{0..7,16..23}; lanes 16-31 +{8..15,24..31}
    auto loadA = [&](int k) -> v16h {
        v8h alo = *(const v8h*)(xrow + k + hi * 8);
        v8h ahi = *(const v8h*)(xrow + k + 16 + hi * 8);
        v16h a;
#pragma unroll
        for (int j = 0; j < 8; ++j) { a[j] = alo[j]; a[8 + j] = ahi[j]; }
        return a;
    };
    // B fragment: lane holds column N=l16 (+tile*16); lanes 0-15 K=+{0..7,16..23},
    // lanes 16-31 K=+{8..15,24..31} (same k order as A).
    auto loadB = [&](int k0, int t) -> v16h {
        const _Float16* wp = &Wl[(t * 16 + l16) * PITCH + k0 + hi * 8];
        v8h blo = *(const v8h*)(wp);
        v8h bhi = *(const v8h*)(wp + 16);
        v16h b;
#pragma unroll
        for (int j = 0; j < 8; ++j) { b[j] = blo[j]; b[8 + j] = bhi[j]; }
        return b;
    };

#pragma unroll 1
    for (int sl = 0; sl < NSLICE; ++sl) {
        const int kbase = sl * KS;
        if (NSLICE > 1 && sl > 0) __syncthreads();   // previous slice fully consumed
        // cooperative fill of the weight slice (all 256 threads, 16B chunks)
        for (int i = threadIdx.x * 8; i < COUT * KS; i += 256 * 8) {
            int c = i / KS, kk = i % KS;             // KS % 8 == 0 -> no row straddle
            *(v8h*)&Wl[c * PITCH + kk] = *(const v8h*)&Wt[(size_t)c * CIN + kbase + kk];
        }
        __syncthreads();

        if (active) {
            v16h a = loadA(kbase);
#pragma unroll 1
            for (int k0 = 0; k0 < KS; k0 += 32) {
                const bool more = (k0 + 32 < KS);    // uniform
                v16h an = a;
                if (more) an = loadA(kbase + k0 + 32);   // prefetch next A chunk

                v16h bcur = loadB(k0, 0);
#pragma unroll
                for (int t = 0; t < TN; ++t) {
                    v16h bnext = bcur;
                    if (t + 1 < TN) bnext = loadB(k0, t + 1);   // load ahead of consume
                    acc[t] = __builtin_amdgcn_wmma_f32_16x16x32_f16(false, a, false, bcur,
                                                                    (short)0, acc[t], false, false);
                    bcur = bnext;
                }
                a = an;
            }
        }
    }

    if (active) {
        // D layout: VGPR r -> row r + 8*(lane/16), col = lane%16 (+tile*16)
#pragma unroll
        for (int t = 0; t < TN; ++t)
#pragma unroll
            for (int r = 0; r < 8; ++r)
                Z[(size_t)(wave * 16 + r + 8 * hi) * COUT + t * 16 + l16] = (_Float16)acc[t][r];
    }
}

// ---------------------------------------------------------------------
// Per-channel batch stats (mean / rsqrt(var+eps)); one block per channel.
__global__ void __launch_bounds__(256)
stats_kernel(const _Float16* __restrict__ Z, float* __restrict__ mean,
             float* __restrict__ rstd, int R, int C)
{
    const int c = blockIdx.x;
    float s = 0.f, s2 = 0.f;
    for (int r = threadIdx.x; r < R; r += 256) {
        float v = (float)Z[(size_t)r * C + c];
        s += v; s2 += v * v;
    }
    __shared__ float sh[256], sh2[256];
    sh[threadIdx.x] = s; sh2[threadIdx.x] = s2;
    __syncthreads();
    for (int sft = 128; sft > 0; sft >>= 1) {
        if (threadIdx.x < sft) { sh[threadIdx.x] += sh[threadIdx.x + sft]; sh2[threadIdx.x] += sh2[threadIdx.x + sft]; }
        __syncthreads();
    }
    if (threadIdx.x == 0) {
        float m   = sh[0] / (float)R;
        float var = sh2[0] / (float)R - m * m;
        mean[c] = m;
        rstd[c] = rsqrtf(var + BN_EPS);
    }
}

// BN + ReLU; output can be strided/offset (used to scatter pos into the 288-ch buffer).
__global__ void __launch_bounds__(256)
bnrelu_kernel(const _Float16* __restrict__ Z, _Float16* __restrict__ Y,
              const float* __restrict__ mean, const float* __restrict__ rstd,
              const float* __restrict__ g, const float* __restrict__ b,
              int R, int C, int ldY, int colOff)
{
    long i = (long)blockIdx.x * blockDim.x + threadIdx.x;
    if (i >= (long)R * C) return;
    int  c = (int)(i % C);
    long r = i / C;
    float v = ((float)Z[i] - mean[c]) * rstd[c] * g[c] + b[c];
    Y[r * ldY + colOff + c] = (_Float16)fmaxf(v, 0.f);
}

// ---------------------------------------------------------------------
__global__ void __launch_bounds__(256)
maxk_pn_kernel(const _Float16* __restrict__ Z, _Float16* __restrict__ xin)
{
    int i = blockIdx.x * blockDim.x + threadIdx.x;   // R_S * 128
    if (i >= R_S * 128) return;
    int c = i % 128, bs = i / 128;
    float m = -3.0e38f;
    for (int k = 0; k < K1_NB; ++k)
        m = fmaxf(m, (float)Z[((size_t)bs * K1_NB + k) * 128 + c]);
    xin[i] = (_Float16)m;
}

// e (R_S x 32) @ w3 (32x3) -> normalize -> dirs row `which`
__global__ void __launch_bounds__(256)
equi_norm_kernel(const _Float16* __restrict__ e, const float* __restrict__ w3,
                 float* __restrict__ dirs, int which)
{
    int r = blockIdx.x * blockDim.x + threadIdx.x;
    if (r >= R_S) return;
    float v0 = 0.f, v1 = 0.f, v2 = 0.f;
#pragma unroll 8
    for (int j = 0; j < 32; ++j) {
        float ej = (float)e[(size_t)r * 32 + j];
        v0 += ej * w3[j * 3 + 0]; v1 += ej * w3[j * 3 + 1]; v2 += ej * w3[j * 3 + 2];
    }
    float n = fmaxf(sqrtf(v0 * v0 + v1 * v1 + v2 * v2), 1e-12f);
    float* d = dirs + (size_t)r * 9 + which * 3;
    d[0] = v0 / n; d[1] = v1 / n; d[2] = v2 / n;
}

__global__ void __launch_bounds__(256)
cross_kernel(float* __restrict__ dirs)
{
    int r = blockIdx.x * blockDim.x + threadIdx.x;
    if (r >= R_S) return;
    float* d = dirs + (size_t)r * 9;
    d[6] = d[1] * d[5] - d[2] * d[4];
    d[7] = d[2] * d[3] - d[0] * d[5];
    d[8] = d[0] * d[4] - d[1] * d[3];
}

// ---------------------------------------------------------------------
// Build g: Xil[:, 0:128]=nb_f-ctr_f, [128:256]=ctr_f; gx = [dots(9), proj_c(3), proj_n(3), dist]
__global__ void __launch_bounds__(256)
gbuild_kernel(const float* __restrict__ nx, const int* __restrict__ knn2,
              const _Float16* __restrict__ f, const float* __restrict__ dirs,
              _Float16* __restrict__ Xil, float* __restrict__ gx)
{
    int t = blockIdx.x * blockDim.x + threadIdx.x;   // (b*S+s)*K2 + k
    if (t >= R_G) return;
    int bs = t / K2_NB;
    int b  = bs / S_PTS;
    int nb = b * S_PTS + knn2[t];
    const float* pc = nx + (size_t)bs * 3;
    const float* pn = nx + (size_t)nb * 3;
    float r0 = pn[0] - pc[0], r1 = pn[1] - pc[1], r2 = pn[2] - pc[2];
    const float* dc = dirs + (size_t)bs * 9;
    const float* dn = dirs + (size_t)nb * 9;

    _Float16* xr = Xil + (size_t)t * 288;
    const _Float16* fc = f + (size_t)bs * 128;
    const _Float16* fn = f + (size_t)nb * 128;
#pragma unroll 4
    for (int c = 0; c < 128; ++c) {
        float a = (float)fc[c], nbv = (float)fn[c];
        xr[c]       = (_Float16)(nbv - a);
        xr[128 + c] = (_Float16)a;
    }
    float* g = gx + (size_t)t * 16;
#pragma unroll
    for (int i = 0; i < 3; ++i)
#pragma unroll
        for (int j = 0; j < 3; ++j)
            g[i * 3 + j] = dc[i * 3 + 0] * dn[j * 3 + 0] + dc[i * 3 + 1] * dn[j * 3 + 1]
                         + dc[i * 3 + 2] * dn[j * 3 + 2];
#pragma unroll
    for (int d = 0; d < 3; ++d)
        g[9 + d] = r0 * dc[d * 3 + 0] + r1 * dc[d * 3 + 1] + r2 * dc[d * 3 + 2];
#pragma unroll
    for (int d = 0; d < 3; ++d)
        g[12 + d] = r0 * dn[d * 3 + 0] + r1 * dn[d * 3 + 1] + r2 * dn[d * 3 + 2];
    g[15] = sqrtf(r0 * r0 + r1 * r1 + r2 * r2);
}

// pos: (R_G x 16) @ (16 x 32) scalar GEMM
__global__ void __launch_bounds__(256)
pos_gemm_kernel(const float* __restrict__ gx, const float* __restrict__ w,
                _Float16* __restrict__ Zp)
{
    int t = blockIdx.x * blockDim.x + threadIdx.x;
    if (t >= R_G) return;
    const float* g = gx + (size_t)t * 16;
    _Float16* zr = Zp + (size_t)t * 32;
#pragma unroll 4
    for (int c = 0; c < 32; ++c) {
        float z = 0.f;
#pragma unroll
        for (int j = 0; j < 16; ++j) z += g[j] * w[j * 32 + c];
        zr[c] = (_Float16)z;
    }
}

// max over K2, transpose to (B, 128, S)
__global__ void __launch_bounds__(256)
final_max_kernel(const _Float16* __restrict__ Z, float* __restrict__ out)
{
    int i = blockIdx.x * blockDim.x + threadIdx.x;   // R_S * 128
    if (i >= R_S * 128) return;
    int c = i % 128, bs = i / 128, b = bs / S_PTS, s = bs % S_PTS;
    float m = -3.0e38f;
    for (int k = 0; k < K2_NB; ++k)
        m = fmaxf(m, (float)Z[((size_t)bs * K2_NB + k) * 128 + c]);
    out[((size_t)b * 128 + c) * S_PTS + s] = m;
}

// =====================================================================
// Host side
// =====================================================================
enum {
    IN_XYZ = 0,
    IN_PN_W1, IN_PN1_G, IN_PN1_B,
    IN_PN_W2, IN_PN2_G, IN_PN2_B,
    IN_PN_W3, IN_PN3_G, IN_PN3_B,
    IN_INV_W1, IN_INV1_G, IN_INV1_B,
    IN_INV_W2, IN_INV2_G, IN_INV2_B,
    IN_EQ1_W1, IN_EQ1A_G, IN_EQ1A_B,
    IN_EQ1_W2, IN_EQ1B_G, IN_EQ1B_B,
    IN_EQ1_W3,
    IN_EQ2_W1, IN_EQ2A_G, IN_EQ2A_B,
    IN_EQ2_W2, IN_EQ2B_G, IN_EQ2B_B,
    IN_EQ2_W3,
    IN_POS_W, IN_POS_G, IN_POS_B,
    IN_IL_W1, IN_IL1_G, IN_IL1_B,
    IN_IL_W2, IN_IL2_G, IN_IL2_B,
    IN_IL_W3, IN_IL3_G, IN_IL3_B
};

static inline size_t alignUp(size_t x, size_t a) { return (x + a - 1) / a * a; }
static inline int cdiv(long n, int b) { return (int)((n + b - 1) / b); }

template <int CIN, int COUT>
static void run_layer(const _Float16* X, const _Float16* Wt, _Float16* Z, int R,
                      const float* g, const float* b,
                      float* mean, float* rstd,
                      _Float16* Y, int ldY, int colOff, hipStream_t st)
{
    wmma_gemm_f16<CIN, COUT><<<cdiv(R / 16, 8), 256, 0, st>>>(X, Wt, Z, R);
    stats_kernel<<<COUT, 256, 0, st>>>(Z, mean, rstd, R, COUT);
    bnrelu_kernel<<<cdiv((long)R * COUT, 256), 256, 0, st>>>(Z, Y, mean, rstd, g, b, R, COUT, ldY, colOff);
}

extern "C" void kernel_launch(void* const* d_in, const int* in_sizes, int n_in,
                              void* d_out, int out_size, void* d_ws, size_t ws_size,
                              hipStream_t stream)
{
    (void)in_sizes; (void)n_in; (void)out_size; (void)ws_size;
    const float* xyz = (const float*)d_in[IN_XYZ];
    auto F = [&](int i) { return (const float*)d_in[i]; };

    float* out      = (float*)d_out;                    // (B,128,S) then new_xyz (B,S,3)
    float* out_nx   = out + (size_t)B_SZ * 128 * S_PTS;

    // ---------------- workspace carve-out ----------------
    char*  w   = (char*)d_ws;
    size_t off = 0;
    auto alloc = [&](size_t bytes) -> void* { void* p = w + off; off = alignUp(off + bytes, 256); return p; };

    int*      fps_idx = (int*)alloc((size_t)B_SZ * S_PTS * 4);
    int*      knn1    = (int*)alloc((size_t)R_PN * 4);
    int*      knn2    = (int*)alloc((size_t)R_G * 4);
    float*    nx      = (float*)alloc((size_t)R_S * 3 * 4);
    float*    dirs    = (float*)alloc((size_t)R_S * 9 * 4);
    float*    gx      = (float*)alloc((size_t)R_G * 16 * 4);
    float*    mean    = (float*)alloc(512 * 4);
    float*    rstd    = (float*)alloc(512 * 4);

    _Float16* w_pn2  = (_Float16*)alloc(64 * 128 * 2);
    _Float16* w_pn3  = (_Float16*)alloc(128 * 128 * 2);
    _Float16* w_inv1 = (_Float16*)alloc(128 * 128 * 2);
    _Float16* w_inv2 = (_Float16*)alloc(128 * 128 * 2);
    _Float16* w_eq1a = (_Float16*)alloc(128 * 64 * 2);
    _Float16* w_eq1b = (_Float16*)alloc(64 * 32 * 2);
    _Float16* w_eq2a = (_Float16*)alloc(128 * 64 * 2);
    _Float16* w_eq2b = (_Float16*)alloc(64 * 32 * 2);
    _Float16* w_il1  = (_Float16*)alloc(288 * 128 * 2);
    _Float16* w_il2  = (_Float16*)alloc(128 * 128 * 2);
    _Float16* w_il3  = (_Float16*)alloc(128 * 128 * 2);

    _Float16* x_inter = (_Float16*)alloc((size_t)R_S * 128 * 2);
    _Float16* s1      = (_Float16*)alloc((size_t)R_S * 128 * 2);
    _Float16* s2      = (_Float16*)alloc((size_t)R_S * 128 * 2);
    _Float16* f_buf   = (_Float16*)alloc((size_t)R_S * 128 * 2);
    _Float16* e_a     = (_Float16*)alloc((size_t)R_S * 64 * 2);
    _Float16* e_b     = (_Float16*)alloc((size_t)R_S * 32 * 2);
    _Float16* z_pos   = (_Float16*)alloc((size_t)R_G * 32 * 2);
    _Float16* bufA    = (_Float16*)alloc((size_t)R_G * 288 * 2);   // >= R_PN*128
    _Float16* bufB    = (_Float16*)alloc((size_t)R_PN * 128 * 2);

    // ---------------- stage 0: FPS + KNN + weight cvt/transpose ----------------
    fps_kernel<<<B_SZ, 256, 0, stream>>>(xyz, fps_idx, nx, out_nx);
    knn_kernel<K1_NB><<<cdiv(R_S, 256), 256, 0, stream>>>(xyz, nx, knn1, N_PTS, S_PTS);

    cvt_transpose_kernel<<<cdiv(64 * 128, 256), 256, 0, stream>>>(F(IN_PN_W2), w_pn2, 64, 128);
    cvt_transpose_kernel<<<cdiv(128 * 128, 256), 256, 0, stream>>>(F(IN_PN_W3), w_pn3, 128, 128);
    cvt_transpose_kernel<<<cdiv(128 * 128, 256), 256, 0, stream>>>(F(IN_INV_W1), w_inv1, 128, 128);
    cvt_transpose_kernel<<<cdiv(128 * 128, 256), 256, 0, stream>>>(F(IN_INV_W2), w_inv2, 128, 128);
    cvt_transpose_kernel<<<cdiv(128 * 64, 256), 256, 0, stream>>>(F(IN_EQ1_W1), w_eq1a, 128, 64);
    cvt_transpose_kernel<<<cdiv(64 * 32, 256), 256, 0, stream>>>(F(IN_EQ1_W2), w_eq1b, 64, 32);
    cvt_transpose_kernel<<<cdiv(128 * 64, 256), 256, 0, stream>>>(F(IN_EQ2_W1), w_eq2a, 128, 64);
    cvt_transpose_kernel<<<cdiv(64 * 32, 256), 256, 0, stream>>>(F(IN_EQ2_W2), w_eq2b, 64, 32);
    cvt_transpose_kernel<<<cdiv(288 * 128, 256), 256, 0, stream>>>(F(IN_IL_W1), w_il1, 288, 128);
    cvt_transpose_kernel<<<cdiv(128 * 128, 256), 256, 0, stream>>>(F(IN_IL_W2), w_il2, 128, 128);
    cvt_transpose_kernel<<<cdiv(128 * 128, 256), 256, 0, stream>>>(F(IN_IL_W3), w_il3, 128, 128);

    // ---------------- stage 1: grouped PointNet MLP ----------------
    group_pn1_kernel<<<cdiv(R_PN, 256), 256, 0, stream>>>(xyz, nx, knn1, F(IN_PN_W1), bufB);
    stats_kernel<<<64, 256, 0, stream>>>(bufB, mean, rstd, R_PN, 64);
    bnrelu_kernel<<<cdiv((long)R_PN * 64, 256), 256, 0, stream>>>(bufB, bufB, mean, rstd,
        F(IN_PN1_G), F(IN_PN1_B), R_PN, 64, 64, 0);

    run_layer<64, 128>(bufB, w_pn2, bufA, R_PN, F(IN_PN2_G), F(IN_PN2_B), mean, rstd, bufA, 128, 0, stream);
    run_layer<128, 128>(bufA, w_pn3, bufB, R_PN, F(IN_PN3_G), F(IN_PN3_B), mean, rstd, bufB, 128, 0, stream);
    maxk_pn_kernel<<<cdiv((long)R_S * 128, 256), 256, 0, stream>>>(bufB, x_inter);

    // invariant branch -> f
    run_layer<128, 128>(x_inter, w_inv1, s1, R_S, F(IN_INV1_G), F(IN_INV1_B), mean, rstd, s1, 128, 0, stream);
    run_layer<128, 128>(s1, w_inv2, s2, R_S, F(IN_INV2_G), F(IN_INV2_B), mean, rstd, f_buf, 128, 0, stream);

    // equivariant branches -> dirs
    run_layer<128, 64>(x_inter, w_eq1a, e_a, R_S, F(IN_EQ1A_G), F(IN_EQ1A_B), mean, rstd, e_a, 64, 0, stream);
    run_layer<64, 32>(e_a, w_eq1b, e_b, R_S, F(IN_EQ1B_G), F(IN_EQ1B_B), mean, rstd, e_b, 32, 0, stream);
    equi_norm_kernel<<<cdiv(R_S, 256), 256, 0, stream>>>(e_b, F(IN_EQ1_W3), dirs, 0);

    run_layer<128, 64>(x_inter, w_eq2a, e_a, R_S, F(IN_EQ2A_G), F(IN_EQ2A_B), mean, rstd, e_a, 64, 0, stream);
    run_layer<64, 32>(e_a, w_eq2b, e_b, R_S, F(IN_EQ2B_G), F(IN_EQ2B_B), mean, rstd, e_b, 32, 0, stream);
    equi_norm_kernel<<<cdiv(R_S, 256), 256, 0, stream>>>(e_b, F(IN_EQ2_W3), dirs, 1);
    cross_kernel<<<cdiv(R_S, 256), 256, 0, stream>>>(dirs);

    // ---------------- stage 2: neighborhood feature mixing ----------------
    knn_kernel<K2_NB><<<cdiv(R_S, 256), 256, 0, stream>>>(nx, nx, knn2, S_PTS, S_PTS);
    gbuild_kernel<<<cdiv(R_G, 256), 256, 0, stream>>>(nx, knn2, f_buf, dirs, bufA, gx);

    pos_gemm_kernel<<<cdiv(R_G, 256), 256, 0, stream>>>(gx, F(IN_POS_W), z_pos);
    stats_kernel<<<32, 256, 0, stream>>>(z_pos, mean, rstd, R_G, 32);
    bnrelu_kernel<<<cdiv((long)R_G * 32, 256), 256, 0, stream>>>(z_pos, bufA, mean, rstd,
        F(IN_POS_G), F(IN_POS_B), R_G, 32, 288, 256);   // scatter into cols 256..287

    run_layer<288, 128>(bufA, w_il1, bufB, R_G, F(IN_IL1_G), F(IN_IL1_B), mean, rstd, bufB, 128, 0, stream);
    run_layer<128, 128>(bufB, w_il2, bufA, R_G, F(IN_IL2_G), F(IN_IL2_B), mean, rstd, bufA, 128, 0, stream);
    run_layer<128, 128>(bufA, w_il3, bufB, R_G, F(IN_IL3_G), F(IN_IL3_B), mean, rstd, bufB, 128, 0, stream);

    final_max_kernel<<<cdiv((long)R_S * 128, 256), 256, 0, stream>>>(bufB, out);
}
